// VimBridge_26439818674667
// MI455X (gfx1250) — matrix operand order repaired and checked by reading, not compiled, not run on hardware
//
#include <hip/hip_runtime.h>

// ---------------------------------------------------------------------------
// Vim bridge block for gfx1250 (MI455X): bf16 WMMA GEMMs (register-blocked
// 16x64 per wave, explicit 2-stage software pipeline over K) +
// register-resident selective scan. wave32.
// ---------------------------------------------------------------------------

#define Dm    1024
#define DINc  2048
#define Nst   16
#define Rr    64
#define DCONVc 4
#define NQc   32
#define Bb    4
#define LIN   2048
#define LSEQ  2080          // LIN + NQ
#define MSEQ  (Bb*LSEQ)     // 8320
#define MIN_  (Bb*LIN)      // 8192

typedef __attribute__((ext_vector_type(16))) __bf16 v16bf;
typedef __attribute__((ext_vector_type(8)))  __bf16 v8bf;
typedef __attribute__((ext_vector_type(8)))  float  v8f;

__device__ __forceinline__ __bf16 f32_to_bf16(float f) {
    union { float f; unsigned u; } v; v.f = f;
    unsigned r = v.u + 0x7FFFu + ((v.u >> 16) & 1u);   // round-to-nearest-even
    union { unsigned short s; __bf16 b; } o;
    o.s = (unsigned short)(r >> 16);
    return o.b;
}
__device__ __forceinline__ float sigmoid_f(float x) { return 1.f / (1.f + __expf(-x)); }
__device__ __forceinline__ float silu_f(float x)    { return x * sigmoid_f(x); }
__device__ __forceinline__ float softplus_f(float x){ return (x > 20.f) ? x : log1pf(__expf(x)); }

// ---------------------------------------------------------------------------
// Pack weight W[K,N] (f32, row-major) into per-tile WMMA B fragments:
// tile index = nt*(K/32)+kt ; within a tile: lane*16+i, where element i of
// lane l is B[kt*32 + (l>>4)*16 + i, nt*16 + (l&15)]  (contiguous k order, same as the A loads).
// ---------------------------------------------------------------------------
__global__ __launch_bounds__(256) void pack_w_kernel(const float* __restrict__ W,
                                                     __bf16* __restrict__ out,
                                                     int K, int N) {
    int tid  = blockIdx.x * 256 + threadIdx.x;
    int lane = tid & 31;
    int tile = tid >> 5;
    int ktiles = K >> 5, ntiles = N >> 4;
    if (tile >= ktiles * ntiles) return;
    int nt = tile / ktiles, kt = tile % ktiles;
    int n  = nt * 16 + (lane & 15);
    int kb = kt * 32 + (lane >> 4) * 16;
    __bf16* o = out + (size_t)tile * 512 + (size_t)lane * 16;
#pragma unroll
    for (int i = 0; i < 16; ++i)
        o[i] = f32_to_bf16(W[(size_t)(kb + i) * N + n]);
}

// f32 -> bf16 bulk convert
__global__ __launch_bounds__(256) void cvt_bf16_kernel(const float* __restrict__ in,
                                                       __bf16* __restrict__ out, size_t n) {
    size_t i = (size_t)blockIdx.x * 256 + threadIdx.x;
    if (i < n) out[i] = f32_to_bf16(in[i]);
}

// ---------------------------------------------------------------------------
// WMMA GEMM: C[M,N] = act( A[M,K] @ Bpack + bias )
//   One wave computes a 16 x (16*NTB) strip. Explicit 2-stage software
//   pipeline: while WMMAs of K-step kt retire, the loads of kt+1 are already
//   in flight in a second register generation, so loadcnt waits stay partial.
//   EPI: 0 = none, 1 = +bias, 2 = softplus(+bias)
//   A row remap: arow = (m/segOut)*segIn + (m%segOut)
// ---------------------------------------------------------------------------
#define GLOAD(s, kt)                                                          \
    do {                                                                      \
        alo[s] = *(const v8bf*)(Abase + (size_t)(kt) * 32);                   \
        ahi[s] = *(const v8bf*)(Abase + (size_t)(kt) * 32 + 8);               \
        _Pragma("unroll")                                                     \
        for (int j = 0; j < NTB; ++j) {                                       \
            const __bf16* bp = Bbase + ((size_t)j * ktiles + (kt)) * 512;     \
            blo[s][j] = *(const v8bf*)(bp);                                   \
            bhi[s][j] = *(const v8bf*)(bp + 8);                               \
        }                                                                     \
    } while (0)

#define GCOMP(s)                                                              \
    do {                                                                      \
        v16bf a;                                                              \
        _Pragma("unroll")                                                     \
        for (int i = 0; i < 8; ++i) { a[i] = alo[s][i]; a[i+8] = ahi[s][i]; } \
        _Pragma("unroll")                                                     \
        for (int j = 0; j < NTB; ++j) {                                       \
            v16bf b;                                                          \
            _Pragma("unroll")                                                 \
            for (int i = 0; i < 8; ++i) {                                     \
                b[i] = blo[s][j][i]; b[i+8] = bhi[s][j][i];                   \
            }                                                                 \
            acc[j] = __builtin_amdgcn_wmma_f32_16x16x32_bf16(                 \
                         false, a, false, b, (short)0, acc[j], false, false); \
        }                                                                     \
    } while (0)

template <int EPI, int NTB>
__global__ __launch_bounds__(256) void wmma_gemm_kernel(const __bf16* __restrict__ A, int lda,
                                                        const __bf16* __restrict__ Bp,
                                                        const float* __restrict__ bias,
                                                        float* __restrict__ C,
                                                        int M, int N, int K,
                                                        int segOut, int segIn) {
    const int lane   = threadIdx.x & 31;
    const int wave   = threadIdx.x >> 5;
    const int ntiles = N >> 4;
    const int ngrp   = ntiles / NTB;
    const int ktiles = K >> 5;
    const int grp    = blockIdx.x * 8 + wave;
    if (grp >= (M >> 4) * ngrp) return;
    const int tm = grp / ngrp;
    const int tg = grp % ngrp;

    // A fragment addressing: this lane feeds row m = tm*16 + (lane&15),
    // K-slab base (lane>>4)*16, second half at +8.
    int am   = tm * 16 + (lane & 15);
    int arow = (am / segOut) * segIn + (am % segOut);
    const __bf16* __restrict__ Abase = A + (size_t)arow * lda + ((lane >> 4) * 16);
    const __bf16* __restrict__ Bbase = Bp + ((size_t)(tg * NTB) * ktiles) * 512
                                          + (size_t)lane * 16;

    v8f zero = {};
    v8f acc[NTB];
#pragma unroll
    for (int j = 0; j < NTB; ++j) acc[j] = zero;

    v8bf alo[2], ahi[2], blo[2][NTB], bhi[2][NTB];

    GLOAD(0, 0);
    int kt = 0;
    for (; kt + 2 <= ktiles; kt += 2) {
        GLOAD(1, kt + 1);
        if (kt + 8 < ktiles) {
            __builtin_prefetch(Abase + (size_t)(kt + 8) * 32, 0, 1);
            __builtin_prefetch(Bbase + (size_t)(kt + 8) * 512, 0, 1);
        }
        GCOMP(0);                       // consumes kt   (loads issued last iter)
        if (kt + 2 < ktiles) GLOAD(0, kt + 2);
        GCOMP(1);                       // consumes kt+1 (loads issued above)
    }
    if (kt < ktiles) GCOMP(0);          // odd-ktiles tail

#pragma unroll
    for (int j = 0; j < NTB; ++j) {
        const int n  = (tg * NTB + j) * 16 + (lane & 15);
        const float bv = (EPI >= 1) ? bias[n] : 0.f;
#pragma unroll
        for (int r = 0; r < 8; ++r) {
            int m = tm * 16 + (lane >> 4) * 8 + r;
            float v = acc[j][r] + bv;
            if (EPI == 2) v = softplus_f(v);
            C[(size_t)m * N + n] = v;
        }
    }
}

// ---------------------------------------------------------------------------
// LayerNorm over D=1024 per row; block per row. Writes fp32 (row remapped by
// seg/segStride) and bf16 at the same remapped row.
// ---------------------------------------------------------------------------
__global__ __launch_bounds__(256) void layernorm_kernel(const float* __restrict__ in,
                                                        const float* __restrict__ w,
                                                        const float* __restrict__ b,
                                                        float* __restrict__ outF,
                                                        __bf16* __restrict__ outB,
                                                        int seg, int segStride) {
    const int row  = blockIdx.x;
    const int orow = (row / seg) * segStride + (row % seg);
    const int tid  = threadIdx.x;
    __shared__ float s1[256], s2[256];
    float a = 0.f, q = 0.f;
    for (int i = tid; i < Dm; i += 256) {
        float v = in[(size_t)row * Dm + i];
        a += v; q += v * v;
    }
    s1[tid] = a; s2[tid] = q; __syncthreads();
    for (int off = 128; off > 0; off >>= 1) {
        if (tid < off) { s1[tid] += s1[tid + off]; s2[tid] += s2[tid + off]; }
        __syncthreads();
    }
    const float mean = s1[0] * (1.f / Dm);
    const float var  = s2[0] * (1.f / Dm) - mean * mean;
    const float inv  = rsqrtf(var + 1e-5f);
    for (int i = tid; i < Dm; i += 256) {
        float v = (in[(size_t)row * Dm + i] - mean) * inv * w[i] + b[i];
        outF[(size_t)orow * Dm + i] = v;
        outB[(size_t)orow * Dm + i] = f32_to_bf16(v);
    }
}

// Residual + LN: LN(seq + mout) -> bf16 into cat buffer (row stride 2*Dm) at colOff.
__global__ __launch_bounds__(256) void resid_ln_kernel(const float* __restrict__ seqF,
                                                       const float* __restrict__ mout,
                                                       const float* __restrict__ w,
                                                       const float* __restrict__ b,
                                                       __bf16* __restrict__ cat,
                                                       int colOff) {
    const int row = blockIdx.x;
    const int tid = threadIdx.x;
    __shared__ float s1[256], s2[256];
    float a = 0.f, q = 0.f;
    for (int i = tid; i < Dm; i += 256) {
        float v = seqF[(size_t)row * Dm + i] + mout[(size_t)row * Dm + i];
        a += v; q += v * v;
    }
    s1[tid] = a; s2[tid] = q; __syncthreads();
    for (int off = 128; off > 0; off >>= 1) {
        if (tid < off) { s1[tid] += s1[tid + off]; s2[tid] += s2[tid + off]; }
        __syncthreads();
    }
    const float mean = s1[0] * (1.f / Dm);
    const float var  = s2[0] * (1.f / Dm) - mean * mean;
    const float inv  = rsqrtf(var + 1e-5f);
    for (int i = tid; i < Dm; i += 256) {
        float v = seqF[(size_t)row * Dm + i] + mout[(size_t)row * Dm + i];
        v = (v - mean) * inv * w[i] + b[i];
        cat[(size_t)row * (2 * Dm) + colOff + i] = f32_to_bf16(v);
    }
}

// Broadcast query tokens into seq rows [LIN, LSEQ) per batch (fp32 + bf16).
__global__ __launch_bounds__(256) void fill_queries_kernel(const float* __restrict__ q,
                                                           float* __restrict__ seqF,
                                                           __bf16* __restrict__ seqB) {
    int idx = blockIdx.x * 256 + threadIdx.x;              // Bb*NQ*Dm
    if (idx >= Bb * NQc * Dm) return;
    int d = idx % Dm;
    int i = (idx / Dm) % NQc;
    int bq = idx / (Dm * NQc);
    float v = q[(size_t)i * Dm + d];
    size_t row = (size_t)bq * LSEQ + LIN + i;
    seqF[row * Dm + d] = v;
    seqB[row * Dm + d] = f32_to_bf16(v);
}

// Depthwise causal (rev=0) / anti-causal (rev=1) conv (k=4) + bias + SiLU.
// Reads xc half of xz [MSEQ, 2*DIN]; writes xc fp32 + bf16 [MSEQ, DIN].
__global__ __launch_bounds__(256) void conv_silu_kernel(const float* __restrict__ xz,
                                                        const float* __restrict__ cw,
                                                        const float* __restrict__ cb,
                                                        float* __restrict__ xc,
                                                        __bf16* __restrict__ xcb,
                                                        int rev) {
    size_t idx = (size_t)blockIdx.x * 256 + threadIdx.x;   // MSEQ*DIN
    if (idx >= (size_t)MSEQ * DINc) return;
    int d = (int)(idx % DINc);
    size_t row = idx / DINc;
    int t = (int)(row % LSEQ);
    int bq = (int)(row / LSEQ);
    float acc = cb[d];
#pragma unroll
    for (int j = 0; j < DCONVc; ++j) {
        int ts = rev ? (t + (DCONVc - 1) - j) : (t - (DCONVc - 1) + j);
        if (ts >= 0 && ts < LSEQ)
            acc += cw[(size_t)j * DINc + d] *
                   xz[((size_t)bq * LSEQ + ts) * (2 * DINc) + d];
    }
    float v = silu_f(acc);
    xc[idx]  = v;
    xcb[idx] = f32_to_bf16(v);
}

// Selective scan: one lane per (b,d), 16 states in registers; B/C staged in LDS;
// next timestep's operands software-prefetched (global_prefetch_b8).
__global__ __launch_bounds__(256) void scan_kernel(const float* __restrict__ xc,
                                                   const float* __restrict__ dt,
                                                   const float* __restrict__ dbc,
                                                   const float* __restrict__ A_log,
                                                   const float* __restrict__ Dp,
                                                   float* __restrict__ y, int rev) {
    int tid = blockIdx.x * 256 + threadIdx.x;              // Bb*DIN lanes
    int d   = tid % DINc;
    int bq  = tid / DINc;                                  // block never straddles b
    float Ad[Nst], h[Nst];
#pragma unroll
    for (int n = 0; n < Nst; ++n) {
        Ad[n] = -__expf(A_log[(size_t)d * Nst + n]);
        h[n]  = 0.f;
    }
    const float Dv = Dp[d];
    __shared__ float sB[Nst], sC[Nst];
    for (int step = 0; step < LSEQ; ++step) {
        int t = rev ? (LSEQ - 1 - step) : step;
        size_t row = (size_t)bq * LSEQ + t;
        if (step + 1 < LSEQ) {
            int t2 = rev ? (LSEQ - 2 - step) : (step + 1);
            size_t row2 = (size_t)bq * LSEQ + t2;
            __builtin_prefetch(dt + row2 * DINc + d, 0, 1);
            __builtin_prefetch(xc + row2 * DINc + d, 0, 1);
        }
        __syncthreads();
        if (threadIdx.x < 32) {
            float v = dbc[row * (Rr + 2 * Nst) + Rr + threadIdx.x];
            if (threadIdx.x < Nst) sB[threadIdx.x] = v;
            else                   sC[threadIdx.x - Nst] = v;
        }
        __syncthreads();
        const float dtv = dt[row * DINc + d];
        const float xv  = xc[row * DINc + d];
        const float dx  = dtv * xv;
        float acc = 0.f;
#pragma unroll
        for (int n = 0; n < Nst; ++n) {
            float dA = __expf(dtv * Ad[n]);
            h[n] = h[n] * dA + dx * sB[n];
            acc += h[n] * sC[n];
        }
        y[row * DINc + d] = acc + xv * Dv;
    }
}

// y *= silu(z); emit bf16 for the W_out GEMM.
__global__ __launch_bounds__(256) void gate_kernel(const float* __restrict__ y,
                                                   const float* __restrict__ xz,
                                                   __bf16* __restrict__ yb) {
    size_t idx = (size_t)blockIdx.x * 256 + threadIdx.x;
    if (idx >= (size_t)MSEQ * DINc) return;
    int d = (int)(idx % DINc);
    size_t row = idx / DINc;
    float z = xz[row * (2 * DINc) + DINc + d];
    yb[idx] = f32_to_bf16(y[idx] * silu_f(z));
}

// ---------------------------------------------------------------------------

static inline int cdiv(long long a, long long b) { return (int)((a + b - 1) / b); }

extern "C" void kernel_launch(void* const* d_in, const int* in_sizes, int n_in,
                              void* d_out, int out_size, void* d_ws, size_t ws_size,
                              hipStream_t stream) {
    // Input order (setup_inputs dict, recursive):
    const float* x1d    = (const float*)d_in[0];
    const float* proj_w = (const float*)d_in[1];
    const float* proj_b = (const float*)d_in[2];
    const float* norm_w = (const float*)d_in[3];
    const float* norm_b = (const float*)d_in[4];
    const float* qtok   = (const float*)d_in[5];
    // fwd: 6..16 ; bwd: 17..27
    const float* out_w  = (const float*)d_in[28];
    const float* out_b  = (const float*)d_in[29];

    char* ws = (char*)d_ws;
    size_t off = 0;
    auto alloc = [&](size_t bytes) -> char* {
        char* p = ws + off;
        off += (bytes + 255) & ~(size_t)255;
        return p;
    };

    __bf16* packProj = (__bf16*)alloc((size_t)Dm * Dm * 2);
    __bf16* packWin[2]  = { (__bf16*)alloc((size_t)Dm * 2 * DINc * 2),
                            (__bf16*)alloc((size_t)Dm * 2 * DINc * 2) };
    __bf16* packWx[2]   = { (__bf16*)alloc((size_t)DINc * (Rr + 2 * Nst) * 2),
                            (__bf16*)alloc((size_t)DINc * (Rr + 2 * Nst) * 2) };
    __bf16* packWdt[2]  = { (__bf16*)alloc((size_t)Rr * DINc * 2),
                            (__bf16*)alloc((size_t)Rr * DINc * 2) };
    __bf16* packWout[2] = { (__bf16*)alloc((size_t)DINc * Dm * 2),
                            (__bf16*)alloc((size_t)DINc * Dm * 2) };
    __bf16* packOutW = (__bf16*)alloc((size_t)2 * DINc * Dm); // 2048*1024 bf16
    __bf16* xinbf  = (__bf16*)alloc((size_t)MIN_ * Dm * 2);
    float*  tmp    = (float*) alloc((size_t)MIN_ * Dm * 4);
    float*  seqF   = (float*) alloc((size_t)MSEQ * Dm * 4);
    __bf16* seqB   = (__bf16*)alloc((size_t)MSEQ * Dm * 2);
    float*  xz     = (float*) alloc((size_t)MSEQ * 2 * DINc * 4);
    float*  xc     = (float*) alloc((size_t)MSEQ * DINc * 4);
    __bf16* xcb    = (__bf16*)alloc((size_t)MSEQ * DINc * 2);
    float*  dbc    = (float*) alloc((size_t)MSEQ * (Rr + 2 * Nst) * 4);
    __bf16* dbcb   = (__bf16*)alloc((size_t)MSEQ * (Rr + 2 * Nst) * 2);
    float*  dtb    = (float*) alloc((size_t)MSEQ * DINc * 4);
    float*  ybuf   = (float*) alloc((size_t)MSEQ * DINc * 4);
    __bf16* yb     = (__bf16*)alloc((size_t)MSEQ * DINc * 2);
    float*  mout   = (float*) alloc((size_t)MSEQ * Dm * 4);
    __bf16* catb   = (__bf16*)alloc((size_t)MSEQ * 2 * Dm * 2);
    (void)ws_size; (void)in_sizes; (void)n_in; (void)out_size;

    auto packW = [&](const float* W, __bf16* dst, int K, int N) {
        long long threads = (long long)(K >> 5) * (N >> 4) * 32;
        pack_w_kernel<<<cdiv(threads, 256), 256, 0, stream>>>(W, dst, K, N);
    };
    auto gemmBlocks = [&](int M, int N, int ntb) {
        return cdiv((long long)(M >> 4) * ((N >> 4) / ntb), 8);
    };

    // ---- weight packing -------------------------------------------------
    packW(proj_w, packProj, Dm, Dm);
    for (int dir = 0; dir < 2; ++dir) {
        const int base = 6 + dir * 11;
        packW((const float*)d_in[base + 0], packWin[dir],  Dm,   2 * DINc);   // W_in
        packW((const float*)d_in[base + 3], packWx[dir],   DINc, Rr + 2*Nst); // W_x
        packW((const float*)d_in[base + 4], packWdt[dir],  Rr,   DINc);       // W_dt
        packW((const float*)d_in[base + 8], packWout[dir], DINc, Dm);         // W_out
    }
    packW(out_w, packOutW, 2 * DINc, Dm);

    // ---- input proj + LN + query concat ---------------------------------
    cvt_bf16_kernel<<<cdiv((long long)MIN_ * Dm, 256), 256, 0, stream>>>(
        x1d, xinbf, (size_t)MIN_ * Dm);
    wmma_gemm_kernel<1, 4><<<gemmBlocks(MIN_, Dm, 4), 256, 0, stream>>>(
        xinbf, Dm, packProj, proj_b, tmp, MIN_, Dm, Dm, MIN_, MIN_);
    layernorm_kernel<<<MIN_, 256, 0, stream>>>(
        tmp, norm_w, norm_b, seqF, seqB, LIN, LSEQ);
    fill_queries_kernel<<<cdiv((long long)Bb * NQc * Dm, 256), 256, 0, stream>>>(
        qtok, seqF, seqB);

    // ---- two directions --------------------------------------------------
    for (int dir = 0; dir < 2; ++dir) {
        const int base = 6 + dir * 11;
        const float* conv_w = (const float*)d_in[base + 1];
        const float* conv_b = (const float*)d_in[base + 2];
        const float* b_dt   = (const float*)d_in[base + 5];
        const float* A_log  = (const float*)d_in[base + 6];
        const float* Dp     = (const float*)d_in[base + 7];
        const float* ln_w   = (const float*)d_in[base + 9];
        const float* ln_b   = (const float*)d_in[base + 10];
        const int rev = dir;  // bwd == anti-causal conv + reverse scan (no flips)

        // xz = seq @ W_in
        wmma_gemm_kernel<0, 4><<<gemmBlocks(MSEQ, 2 * DINc, 4), 256, 0, stream>>>(
            seqB, Dm, packWin[dir], nullptr, xz, MSEQ, 2 * DINc, Dm, MSEQ, MSEQ);
        // xc = silu(conv(xc_half) + b)
        conv_silu_kernel<<<cdiv((long long)MSEQ * DINc, 256), 256, 0, stream>>>(
            xz, conv_w, conv_b, xc, xcb, rev);
        // dbc = xc @ W_x   (N=96 -> 6 n-tiles -> NTB=2)
        wmma_gemm_kernel<0, 2><<<gemmBlocks(MSEQ, Rr + 2 * Nst, 2), 256, 0, stream>>>(
            xcb, DINc, packWx[dir], nullptr, dbc, MSEQ, Rr + 2 * Nst, DINc, MSEQ, MSEQ);
        cvt_bf16_kernel<<<cdiv((long long)MSEQ * (Rr + 2 * Nst), 256), 256, 0, stream>>>(
            dbc, dbcb, (size_t)MSEQ * (Rr + 2 * Nst));
        // dt = softplus(dbc[:, :64] @ W_dt + b_dt)
        wmma_gemm_kernel<2, 4><<<gemmBlocks(MSEQ, DINc, 4), 256, 0, stream>>>(
            dbcb, Rr + 2 * Nst, packWdt[dir], b_dt, dtb, MSEQ, DINc, Rr, MSEQ, MSEQ);
        // selective scan
        scan_kernel<<<(Bb * DINc) / 256, 256, 0, stream>>>(
            xc, dtb, dbc, A_log, Dp, ybuf, rev);
        // y *= silu(z)
        gate_kernel<<<cdiv((long long)MSEQ * DINc, 256), 256, 0, stream>>>(
            ybuf, xz, yb);
        // mamba_out = y @ W_out
        wmma_gemm_kernel<0, 4><<<gemmBlocks(MSEQ, Dm, 4), 256, 0, stream>>>(
            yb, DINc, packWout[dir], nullptr, mout, MSEQ, Dm, DINc, MSEQ, MSEQ);
        // LN(seq + mamba_out) -> cat[:, dir*Dm : dir*Dm+Dm]
        resid_ln_kernel<<<MSEQ, 256, 0, stream>>>(
            seqF, mout, ln_w, ln_b, catb, dir * Dm);
    }

    // ---- final projection, dropping the NQ query rows per batch ----------
    wmma_gemm_kernel<1, 4><<<gemmBlocks(MIN_, Dm, 4), 256, 0, stream>>>(
        catb, 2 * Dm, packOutW, out_b, (float*)d_out, MIN_, Dm, 2 * DINc,
        /*segOut=*/LIN, /*segIn=*/LSEQ);
}
